// embed_net_4475355923013
// MI455X (gfx1250) — matrix operand order repaired and checked by reading, not compiled, not run on hardware
//
#include <hip/hip_runtime.h>
#include <hip/hip_bf16.h>

typedef _Float16 f16;
typedef __attribute__((ext_vector_type(16))) _Float16 v16h;
typedef __attribute__((ext_vector_type(8)))  float    v8f;
typedef __attribute__((ext_vector_type(4)))  unsigned int u32x4;
typedef __attribute__((ext_vector_type(8)))  int i32x8;
typedef __attribute__((ext_vector_type(4)))  int i32x4;

union FragH { v16h v; uint4 u[2]; };

#define HALFB 32
#define BTOT  64
#define CDIM  2048
#define HH    24
#define WW    12
#define HWP   288
#define CQ    512
#define MTOT  1024            // q(512) ++ k(512) output channels
#define KCONV 2048
#define TEMPR 50.0f
#define PIT   40              // LDS k-pitch for 32-wide tiles (80B rows, bank-stride 20)
#define APIT  520             // LDS pitch for 512-wide q-tile (1040B rows)

#define FEATH_ELEMS ((size_t)BTOT * CDIM * HWP)      // 37,748,736
#define WR_ELEMS    ((size_t)9 * MTOT * KCONV)       // 18,874,368
#define CONV_ELEMS  ((size_t)BTOT * MTOT * HWP)      // 18,874,368
#define PR_ELEMS    ((size_t)2 * BTOT * HWP * HWP)   // 10,616,832

#if __has_builtin(__builtin_amdgcn_tensor_load_to_lds) && __has_builtin(__builtin_amdgcn_s_wait_tensorcnt)
#define HAVE_TDM 1
#else
#define HAVE_TDM 0
#endif

#if HAVE_TDM
// Issue a 2D TDM tile load (f16 elements) into LDS with 16B padding after every
// 64B of data (=> 80B LDS row pitch == PIT f16). One call per wave; EXEC ignored.
__device__ __forceinline__ void tdm_load_2d(const void* tile_gaddr, unsigned lds_off,
                                            unsigned long long dim0, unsigned long long dim1,
                                            unsigned tile0, unsigned tile1,
                                            unsigned long long stride0) {
    unsigned long long ga = (unsigned long long)(uintptr_t)tile_gaddr;
    u32x4 g0;
    g0[0] = 1u;                                        // count=1 (valid), no gather
    g0[1] = lds_off;                                   // LDS byte address
    g0[2] = (unsigned)(ga & 0xffffffffu);              // global_addr[31:0]
    g0[3] = (unsigned)((ga >> 32) & 0x1ffffffu) | (2u << 30);  // addr[56:32] | type=2
    i32x8 g1;
    g1[0] = (int)((1u << 16)      // data_size = 2 bytes
                | (1u << 20)      // pad_enable
                | (3u << 22)      // pad_interval: 16 DWORDs (64B)
                | (3u << 25));    // pad_amount:   4 DWORDs (16B)
    g1[1] = (int)((unsigned)(dim0 & 0xffffu) << 16);                       // tensor_dim0[15:0]
    g1[2] = (int)(((unsigned)(dim0 >> 16) & 0xffffu) |
                  ((unsigned)(dim1 & 0xffffu) << 16));                     // dim0[31:16] | dim1[15:0]
    g1[3] = (int)(((unsigned)(dim1 >> 16) & 0xffffu) | (tile0 << 16));     // dim1[31:16] | tile_dim0
    g1[4] = (int)tile1;                                                    // tile_dim1 | tile_dim2=0
    g1[5] = (int)(unsigned)(stride0 & 0xffffffffu);                        // tensor_dim0_stride[31:0]
    g1[6] = (int)(unsigned)((stride0 >> 32) & 0xffffu);                    // stride[47:32]
    g1[7] = 0;
    i32x4 gz = {0, 0, 0, 0};
#if defined(__clang_major__) && (__clang_major__ >= 23)
    i32x8 gz8 = {0, 0, 0, 0, 0, 0, 0, 0};
    __builtin_amdgcn_tensor_load_to_lds(g0, g1, gz, gz, gz8, 0);
#else
    __builtin_amdgcn_tensor_load_to_lds(g0, g1, gz, gz, 0);
#endif
}
#endif

#define WMMA_F16(A, B, C) __builtin_amdgcn_wmma_f32_16x16x32_f16(false, (A), false, (B), (short)0, (C), false, false)

// ---------------------------------------------------------------- utilities
__global__ void k_zero(float* p, int n) {
    int i = blockIdx.x * 256 + threadIdx.x;
    if (i < n) p[i] = 0.0f;
}

__global__ void k_cast_feat(const float* fv, const float* ft, f16* feath) {
    size_t i = (size_t)blockIdx.x * 256 + threadIdx.x;
    if (i >= FEATH_ELEMS) return;
    size_t per = (size_t)CDIM * HWP;
    size_t b = i / per;
    float v = (b < HALFB) ? fv[i] : ft[i - (size_t)HALFB * per];
    feath[i] = (f16)v;
}

// reorder q_w/k_w [512,2048,3,3] f32 -> wr [9][1024][2048] f16 (m: q then k)
__global__ void k_cast_w(const float* qw, const float* kw, f16* wr) {
    size_t i = (size_t)blockIdx.x * 256 + threadIdx.x;
    if (i >= WR_ELEMS) return;
    size_t rs  = i / ((size_t)MTOT * KCONV);
    size_t rem = i % ((size_t)MTOT * KCONV);
    size_t m   = rem / KCONV;
    size_t ci  = rem % KCONV;
    float v;
    if (m < CQ) v = qw[(m * KCONV + ci) * 9 + rs];
    else        v = kw[((m - CQ) * KCONV + ci) * 9 + rs];
    wr[i] = (f16)v;
}

__global__ __launch_bounds__(HWP) void k_mask(const float* fv, const float* ft, float* maskb) {
    int b = blockIdx.x, p = threadIdx.x;
    const float* base = (b < HALFB) ? fv + (size_t)b * CDIM * HWP
                                    : ft + (size_t)(b - HALFB) * CDIM * HWP;
    float s = 0.f;
    for (int c = 0; c < CDIM; ++c) { float x = base[(size_t)c * HWP + p]; s += x * x; }
    float n = sqrtf(s);
    __shared__ float sm[HWP];
    __shared__ float mn, mx;
    sm[p] = n; __syncthreads();
    if (p == 0) {
        float a = sm[0], c = sm[0];
        for (int i = 1; i < HWP; ++i) { a = fminf(a, sm[i]); c = fmaxf(c, sm[i]); }
        mn = a; mx = c;
    }
    __syncthreads();
    maskb[b * HWP + p] = (n - mn) / ((mx - mn) + 1e-12f);
}

__global__ __launch_bounds__(HWP) void k_minmax(const float* in, float* out) {
    int b = blockIdx.x, p = threadIdx.x;
    float n = in[b * HWP + p];
    __shared__ float sm[HWP];
    __shared__ float mn, mx;
    sm[p] = n; __syncthreads();
    if (p == 0) {
        float a = sm[0], c = sm[0];
        for (int i = 1; i < HWP; ++i) { a = fminf(a, sm[i]); c = fmaxf(c, sm[i]); }
        mn = a; mx = c;
    }
    __syncthreads();
    out[b * HWP + p] = (n - mn) / ((mx - mn) + 1e-12f);
}

// ------------------------------------------------- conv3x3 as implicit GEMM
// out[b][m][p] = sum_{rs,ci} wr[rs][m][ci] * feath[b][ci][shift(p,rs)] + bias
// WG: 256 thr (8 waves), tile 128M x 48N, K-tile 32, 9 shifts.
// Double-buffered: TDM(A) + im2col(B) for tile it+1 overlap WMMAs of tile it.
__global__ __launch_bounds__(256) void k_conv(const f16* __restrict__ feath,
                                              const f16* __restrict__ wr,
                                              const float* qb, const float* kb,
                                              f16* __restrict__ conv) {
    __shared__ __align__(16) f16 Al[2][128][PIT];
    __shared__ __align__(16) f16 Bl[2][48][PIT];
    int nb = blockIdx.x, mb = blockIdx.y, b = blockIdx.z;
    int t = threadIdx.x, lane = t & 31, wid = t >> 5;
    int hi = lane >> 4, lm = lane & 15;
    int m0 = mb * 128, n0 = nb * 48;
    v8f acc[3] = {};
    const f16* fb = feath + (size_t)b * CDIM * HWP;

    int bn[6], bkk[6];
    for (int i = 0; i < 6; ++i) { int e = t + i * 256; bkk[i] = e / 48; bn[i] = e % 48; }
    int  boff[6];
    bool bok[6];
    int  staged_rs = -1;
#if HAVE_TDM
    unsigned lds_al[2] = { (unsigned)(uintptr_t)&Al[0][0][0], (unsigned)(uintptr_t)&Al[1][0][0] };
#endif

    auto stage = [&](int it, int buf) {
        int rs = it >> 6, kt = it & 63;
        if (rs != staged_rs) {          // 9 transitions out of 576 iterations
            staged_rs = rs;
            int dy = rs / 3 - 1, dx = rs % 3 - 1;
            for (int i = 0; i < 6; ++i) {
                int p = n0 + bn[i];
                int y = p / WW + dy, x = p % WW + dx;
                bool ok = ((unsigned)y < HH) && ((unsigned)x < WW);
                bok[i]  = ok;
                boff[i] = ok ? (bkk[i] * HWP + y * WW + x) : 0;
            }
        }
        const f16* asrc = wr + ((size_t)rs * MTOT + m0) * KCONV + (size_t)kt * 32;
#if HAVE_TDM
        if (wid == 0)
            tdm_load_2d(asrc, lds_al[buf], KCONV, (unsigned long long)9 * MTOT, 32, 128, KCONV);
#else
        {   int row = t >> 1, ch = (t & 1) * 16;
            const uint4* g = reinterpret_cast<const uint4*>(asrc + (size_t)row * KCONV + ch);
            uint4 x0 = g[0], x1 = g[1];
            uint4* d = reinterpret_cast<uint4*>(&Al[buf][row][ch]);
            d[0] = x0; d[1] = x1;
        }
#endif
        const f16* fbk = fb + (size_t)kt * 32 * HWP;
        for (int i = 0; i < 6; ++i) {               // branch-free im2col (clamp + select)
            f16 v = fbk[boff[i]];
            Bl[buf][bn[i]][bkk[i]] = bok[i] ? v : (f16)0.f;
        }
    };

    stage(0, 0);
#if HAVE_TDM
    if (wid == 0) __builtin_amdgcn_s_wait_tensorcnt(0);
#endif
    __syncthreads();

    const int NIT = 9 * (KCONV / 32);
    for (int it = 0; it < NIT; ++it) {
        int cur = it & 1, nxt = cur ^ 1;
        if (it + 1 < NIT) stage(it + 1, nxt);       // overlaps compute below
        FragH a, b0, b1, b2;
        { const uint4* s0 = reinterpret_cast<const uint4*>(&Al[cur][wid * 16 + lm][16 * hi]);
          const uint4* s1 = reinterpret_cast<const uint4*>(&Al[cur][wid * 16 + lm][16 * hi + 8]);
          a.u[0] = s0[0]; a.u[1] = s1[0]; }
        { const uint4* s = reinterpret_cast<const uint4*>(&Bl[cur][lm][16 * hi]);
          b0.u[0] = s[0]; b0.u[1] = s[1]; }
        { const uint4* s = reinterpret_cast<const uint4*>(&Bl[cur][16 + lm][16 * hi]);
          b1.u[0] = s[0]; b1.u[1] = s[1]; }
        { const uint4* s = reinterpret_cast<const uint4*>(&Bl[cur][32 + lm][16 * hi]);
          b2.u[0] = s[0]; b2.u[1] = s[1]; }
        acc[0] = WMMA_F16(a.v, b0.v, acc[0]);
        acc[1] = WMMA_F16(a.v, b1.v, acc[1]);
        acc[2] = WMMA_F16(a.v, b2.v, acc[2]);
#if HAVE_TDM
        if (wid == 0) __builtin_amdgcn_s_wait_tensorcnt(0);
#endif
        __syncthreads();                            // single barrier per K-tile
    }
    const float* bptr = (m0 < CQ) ? qb : (kb - CQ); // m-block never straddles q/k halves
    for (int nt = 0; nt < 3; ++nt) {
        int q = n0 + nt * 16 + lm;
        for (int r = 0; r < 8; ++r) {
            int mg = m0 + wid * 16 + r + 8 * hi;
            conv[(size_t)b * MTOT * HWP + (size_t)mg * HWP + q] = (f16)(acc[nt][r] + bptr[mg]);
        }
    }
}

// l2-normalize fq/fk halves over 512 channels, in place
__global__ __launch_bounds__(HWP) void k_norm(f16* conv) {
    int b = blockIdx.x, half = blockIdx.y, p = threadIdx.x;
    f16* base = conv + (size_t)b * MTOT * HWP + (size_t)half * CQ * HWP;
    float s = 0.f;
    for (int c = 0; c < CQ; ++c) { float x = (float)base[(size_t)c * HWP + p]; s += x * x; }
    float inv = 1.0f / fmaxf(sqrtf(s), 1e-12f);
    for (int c = 0; c < CQ; ++c) {
        size_t o = (size_t)c * HWP + p;
        base[o] = (f16)((float)base[o] * inv);
    }
}

// ---------------------------------------- sim GEMM + softmax + comask (fused)
// WG: 288 thr (9 waves). 16 q-rows x 288 k-cols, K=512. pr stored [br][b][q][k] f16.
__global__ __launch_bounds__(HWP) void k_sim(const f16* __restrict__ conv,
                                             const int* pidx, const int* nidx,
                                             const float* maskb,
                                             f16* __restrict__ pr, float* comask_raw) {
    __shared__ __align__(16) f16 Aq[16][APIT];
    __shared__ __align__(16) f16 Bl[HWP][PIT];
    __shared__ float Sld[16][HWP];
    int qb = blockIdx.x, b = blockIdx.y, br = blockIdx.z;
    int t = threadIdx.x, lane = t & 31, wid = t >> 5;
    int hi = lane >> 4, lm = lane & 15;
    int q0 = qb * 16;
    int tb = (br ? nidx : pidx)[b];
    const f16* fq = conv + (size_t)b * MTOT * HWP;                      // fqn: ch 0..511
    const f16* fk = conv + (size_t)tb * MTOT * HWP + (size_t)CQ * HWP;  // fkn: ch 512..1023
    for (int id = t; id < 2 * CQ; id += HWP) {      // Aq[qi][c] = fq[c][q0+qi]
        int c = id >> 1, h2 = id & 1;
        uint4 vsrc = *reinterpret_cast<const uint4*>(fq + (size_t)c * HWP + q0 + 8 * h2);
        const f16* e = reinterpret_cast<const f16*>(&vsrc);
#pragma unroll
        for (int j = 0; j < 8; ++j) Aq[8 * h2 + j][c] = e[j];
    }
    v8f acc[2] = {};
    for (int ct = 0; ct < CQ / 32; ++ct) {
        __syncthreads();
        for (int id = t; id < 32 * (HWP / 8); id += HWP) {  // Bl[k][c] = fk[ct*32+c][k]
            int cc = id / (HWP / 8), kg = id % (HWP / 8);
            uint4 vsrc = *reinterpret_cast<const uint4*>(fk + (size_t)(ct * 32 + cc) * HWP + kg * 8);
            const f16* e = reinterpret_cast<const f16*>(&vsrc);
#pragma unroll
            for (int j = 0; j < 8; ++j) Bl[kg * 8 + j][cc] = e[j];
        }
        __syncthreads();
        FragH a, b0, b1;
        { const uint4* s0 = reinterpret_cast<const uint4*>(&Aq[lm][ct * 32 + 16 * hi]);
          const uint4* s1 = reinterpret_cast<const uint4*>(&Aq[lm][ct * 32 + 16 * hi + 8]);
          a.u[0] = s0[0]; a.u[1] = s1[0]; }
        { const uint4* s = reinterpret_cast<const uint4*>(&Bl[wid * 32 + lm][16 * hi]);
          b0.u[0] = s[0]; b0.u[1] = s[1]; }
        { const uint4* s = reinterpret_cast<const uint4*>(&Bl[wid * 32 + 16 + lm][16 * hi]);
          b1.u[0] = s[0]; b1.u[1] = s[1]; }
        acc[0] = WMMA_F16(a.v, b0.v, acc[0]);
        acc[1] = WMMA_F16(a.v, b1.v, acc[1]);
    }
    __syncthreads();
    for (int nt = 0; nt < 2; ++nt)
        for (int r = 0; r < 8; ++r)
            Sld[r + 8 * hi][wid * 32 + nt * 16 + lm] = acc[nt][r];
    __syncthreads();
    if (t < 16) {   // stable softmax over k with temperature; fused comask dot
        int row = t;
        float m = -1e30f;
        for (int k = 0; k < HWP; ++k) m = fmaxf(m, Sld[row][k]);
        float sum = 0.f;
        for (int k = 0; k < HWP; ++k) { float e2 = expf(TEMPR * (Sld[row][k] - m)); Sld[row][k] = e2; sum += e2; }
        float inv = 1.0f / sum, dot = 0.f;
        const float* mk = maskb + tb * HWP;
        for (int k = 0; k < HWP; ++k) { float pv = Sld[row][k] * inv; Sld[row][k] = pv; dot += pv * mk[k]; }
        if (br == 0) comask_raw[b * HWP + q0 + row] = maskb[b * HWP + q0 + row] * dot;
    }
    __syncthreads();
    f16* prb = pr + ((size_t)br * BTOT + b) * HWP * HWP;
    for (int i = 0; i < 16; ++i)
        prb[(size_t)(q0 + i) * HWP + t] = (f16)Sld[i][t];
}

// --------------------- warp GEMM + recon + distance accumulation (fused)
// warp[c][q] = sum_k feath[tb][c][k] * pr[q][k];  M=2048, N=288, K=288. Double-buffered.
__global__ __launch_bounds__(256) void k_warp(const f16* __restrict__ feath,
                                              const f16* __restrict__ pr,
                                              const int* pidx, const int* nidx,
                                              const float* maskb,
                                              const float* fv, const float* ft,
                                              float* __restrict__ out, float* dsq) {
    __shared__ __align__(16) f16 Al[2][128][PIT];
    __shared__ __align__(16) f16 Bl[2][48][PIT];
    int nb = blockIdx.x, mb = blockIdx.y;
    int bz = blockIdx.z, b = bz & 63, br = bz >> 6;
    int t = threadIdx.x, lane = t & 31, wid = t >> 5;
    int hi = lane >> 4, lm = lane & 15;
    int c0 = mb * 128, q0 = nb * 48;
    int tb = (br ? nidx : pidx)[b];
    const f16* A = feath + (size_t)tb * CDIM * HWP + (size_t)c0 * HWP;
    const f16* B = pr + ((size_t)br * BTOT + b) * HWP * HWP;
    v8f acc[3] = {};
#if HAVE_TDM
    unsigned lds_al[2] = { (unsigned)(uintptr_t)&Al[0][0][0], (unsigned)(uintptr_t)&Al[1][0][0] };
#endif

    auto stage = [&](int kt, int buf) {
#if HAVE_TDM
        if (wid == 0)
            tdm_load_2d(A + (size_t)kt * 32, lds_al[buf], HWP, CDIM, 32, 128, HWP);
#else
        {   int row = t >> 1, ch = (t & 1) * 16;
            const uint4* g = reinterpret_cast<const uint4*>(A + (size_t)row * HWP + kt * 32 + ch);
            uint4 x0 = g[0], x1 = g[1];
            uint4* d = reinterpret_cast<uint4*>(&Al[buf][row][ch]);
            d[0] = x0; d[1] = x1;
        }
#endif
        if (t < 192) {  // B tile: 48 rows x 32 k, fully vectorized 16B copies
            int n = t >> 2, c16 = (t & 3) * 8;
            uint4 vsrc = *reinterpret_cast<const uint4*>(B + (size_t)(q0 + n) * HWP + kt * 32 + c16);
            *reinterpret_cast<uint4*>(&Bl[buf][n][c16]) = vsrc;
        }
    };

    stage(0, 0);
#if HAVE_TDM
    if (wid == 0) __builtin_amdgcn_s_wait_tensorcnt(0);
#endif
    __syncthreads();

    for (int kt = 0; kt < HWP / 32; ++kt) {
        int cur = kt & 1, nxt = cur ^ 1;
        if (kt + 1 < HWP / 32) stage(kt + 1, nxt);
        FragH a, b0, b1, b2;
        { const uint4* s0 = reinterpret_cast<const uint4*>(&Al[cur][wid * 16 + lm][16 * hi]);
          const uint4* s1 = reinterpret_cast<const uint4*>(&Al[cur][wid * 16 + lm][16 * hi + 8]);
          a.u[0] = s0[0]; a.u[1] = s1[0]; }
        { const uint4* s = reinterpret_cast<const uint4*>(&Bl[cur][lm][16 * hi]);
          b0.u[0] = s[0]; b0.u[1] = s[1]; }
        { const uint4* s = reinterpret_cast<const uint4*>(&Bl[cur][16 + lm][16 * hi]);
          b1.u[0] = s[0]; b1.u[1] = s[1]; }
        { const uint4* s = reinterpret_cast<const uint4*>(&Bl[cur][32 + lm][16 * hi]);
          b2.u[0] = s[0]; b2.u[1] = s[1]; }
        acc[0] = WMMA_F16(a.v, b0.v, acc[0]);
        acc[1] = WMMA_F16(a.v, b1.v, acc[1]);
        acc[2] = WMMA_F16(a.v, b2.v, acc[2]);
#if HAVE_TDM
        if (wid == 0) __builtin_amdgcn_s_wait_tensorcnt(0);
#endif
        __syncthreads();
    }
    const float* fsrc = (b < HALFB) ? fv + (size_t)b * CDIM * HWP
                                    : ft + (size_t)(b - HALFB) * CDIM * HWP;
    for (int nt = 0; nt < 3; ++nt) {
        int q = q0 + nt * 16 + lm;
        float mq = maskb[b * HWP + q];
        float s = 0.f;
        for (int r = 0; r < 8; ++r) {
            int c = c0 + wid * 16 + r + 8 * hi;
            float f = fsrc[(size_t)c * HWP + q];
            float recon = mq * acc[nt][r] + (1.0f - mq) * f;
            if (br == 0) out[(size_t)b * CDIM * HWP + (size_t)c * HWP + q] = recon;
            float d = f - recon + 1e-6f;
            s += d * d;
        }
        atomicAdd(&dsq[(size_t)br * BTOT * HWP + b * HWP + q], s);
    }
}

// triplet loss: mean(comask * relu(sqrt(dsq_pos) - sqrt(dsq_neg) + 0.3))
__global__ __launch_bounds__(256) void k_loss(const float* comask, const float* dsq, float* out_loss) {
    __shared__ float red[256];
    int t = threadIdx.x;
    float s = 0.f;
    for (int e = t; e < BTOT * HWP; e += 256) {
        float dap = sqrtf(dsq[e]);
        float dan = sqrtf(dsq[BTOT * HWP + e]);
        s += comask[e] * fmaxf(dap - dan + 0.3f, 0.f);
    }
    red[t] = s; __syncthreads();
    for (int o = 128; o > 0; o >>= 1) { if (t < o) red[t] += red[t + o]; __syncthreads(); }
    if (t == 0) *out_loss = red[0] / (float)(BTOT * HWP);
}

extern "C" void kernel_launch(void* const* d_in, const int* in_sizes, int n_in,
                              void* d_out, int out_size, void* d_ws, size_t ws_size,
                              hipStream_t stream) {
    (void)in_sizes; (void)n_in; (void)out_size; (void)ws_size;
    const float* fv  = (const float*)d_in[0];
    const float* ft  = (const float*)d_in[1];
    const float* qw  = (const float*)d_in[4];
    const float* qb  = (const float*)d_in[5];
    const float* kw  = (const float*)d_in[6];
    const float* kb  = (const float*)d_in[7];
    const int* pidx  = (const int*)d_in[8];
    const int* nidx  = (const int*)d_in[9];
    float* out = (float*)d_out;

    char* ws = (char*)d_ws;
    size_t off = 0;
    f16* feath = (f16*)(ws + off);      off += FEATH_ELEMS * sizeof(f16);   // 75.5 MB
    f16* wr    = (f16*)(ws + off);      off += WR_ELEMS * sizeof(f16);      // 37.7 MB
    f16* conv  = (f16*)(ws + off);      off += CONV_ELEMS * sizeof(f16);    // 37.7 MB
    f16* pr    = (f16*)(ws + off);      off += PR_ELEMS * sizeof(f16);      // 21.2 MB
    float* maskb   = (float*)(ws + off); off += (size_t)BTOT * HWP * sizeof(float);
    float* comraw  = (float*)(ws + off); off += (size_t)BTOT * HWP * sizeof(float);
    float* comask  = (float*)(ws + off); off += (size_t)BTOT * HWP * sizeof(float);
    float* dsq     = (float*)(ws + off); off += (size_t)2 * BTOT * HWP * sizeof(float);

    int ndsq = 2 * BTOT * HWP;
    k_zero<<<dim3((ndsq + 255) / 256), dim3(256), 0, stream>>>(dsq, ndsq);
    k_cast_feat<<<dim3((unsigned)((FEATH_ELEMS + 255) / 256)), dim3(256), 0, stream>>>(fv, ft, feath);
    k_cast_w<<<dim3((unsigned)((WR_ELEMS + 255) / 256)), dim3(256), 0, stream>>>(qw, kw, wr);
    k_mask<<<dim3(BTOT), dim3(HWP), 0, stream>>>(fv, ft, maskb);
    k_conv<<<dim3(6, 8, BTOT), dim3(256), 0, stream>>>(feath, wr, qb, kb, conv);
    k_norm<<<dim3(BTOT, 2), dim3(HWP), 0, stream>>>(conv);
    k_sim<<<dim3(18, BTOT, 2), dim3(HWP), 0, stream>>>(conv, pidx, nidx, maskb, pr, comraw);
    k_warp<<<dim3(6, 16, 2 * BTOT), dim3(256), 0, stream>>>(feath, pr, pidx, nidx, maskb, fv, ft, out, dsq);
    k_minmax<<<dim3(BTOT), dim3(HWP), 0, stream>>>(comraw, comask);
    k_loss<<<dim3(1), dim3(256), 0, stream>>>(comask, dsq, out + FEATH_ELEMS);
}
